// GQACausalAttention_36240934044058
// MI455X (gfx1250) — matrix operand order repaired and checked by reading, not compiled, not run on hardware
//
#include <hip/hip_runtime.h>
#include <math.h>

// ---------------------------------------------------------------------------
// GQA causal attention for MI455X (gfx1250, wave32, WMMA bf16->f32).
// Pipeline: cvt/transposes -> QKV WMMA GEMMs (TDM-staged B tiles in LDS,
// double buffered) -> hier-RoPE pack -> flash attention (WMMA + online
// softmax, P staged through LDS) -> output WMMA GEMM.
// ---------------------------------------------------------------------------

typedef __bf16 bf16_t;
typedef __attribute__((ext_vector_type(16))) __bf16 v16bf;
typedef __attribute__((ext_vector_type(8)))  __bf16 v8bf;
typedef __attribute__((ext_vector_type(8)))  float  v8f;
typedef unsigned int v4u __attribute__((ext_vector_type(4)));
typedef int          v8i __attribute__((ext_vector_type(8)));
typedef int          v4i __attribute__((ext_vector_type(4)));

#define T_SEQ  2048
#define DMODEL 2048
#define NH     32
#define NKV    8
#define HDIM   64

// Concatenate two 16-byte loads into one 16-element bf16 WMMA fragment.
__device__ __forceinline__ v16bf frag16(const bf16_t* lo, const bf16_t* hi) {
  v8bf a = *(const v8bf*)lo;
  v8bf b = *(const v8bf*)hi;
  return __builtin_shufflevector(a, b, 0, 1, 2, 3, 4, 5, 6, 7,
                                        8, 9, 10, 11, 12, 13, 14, 15);
}

__device__ __forceinline__ v8f wmma_bf16(v16bf a, v16bf b, v8f c) {
  // (neg_a, A, neg_b, B, c_mod, C, reuse_a, reuse_b)
  return __builtin_amdgcn_wmma_f32_16x16x32_bf16(false, a, false, b,
                                                 (short)0, c, false, false);
}

// TDM load: 64 rows x 32 elements of bf16 from BT[row, k0..k0+31] (row stride
// K elements) into LDS at byte offset lds_off, laid out [64][32] contiguous.
// D# per CDNA5 ISA 8.3/8.4: group0 {count=1, lds_addr, global_addr(57b),
// type=2}; group1 {data_size=1(2B), tensor_dim0=K-k0, tensor_dim1=64,
// tile_dim0=32, tile_dim1=64, tensor_dim0_stride=K}. Groups 2/3 (and the
// trailing group on this 6-arg toolchain) are zero: 2-D, no gather/iterate.
__device__ __forceinline__ void tdm_load_b(const bf16_t* BT, int row0, int k0,
                                           int K, unsigned lds_off) {
  unsigned long long ga =
      (unsigned long long)(const void*)(BT + (size_t)row0 * K + k0);
  v4u g0;
  g0[0] = 1u;                                   // count=1, user mode
  g0[1] = lds_off;                              // LDS byte address
  g0[2] = (unsigned)ga;                         // global_addr[31:0]
  g0[3] = (unsigned)((ga >> 32) & 0x01FFFFFFu)  // global_addr[56:32]
          | (2u << 30);                         // type=2 ("image")
  const unsigned td0 = (unsigned)(K - k0);      // tensor_dim0 (elements)
  v8i g1;
  g1[0] = (int)(1u << 16);                      // data_size=1 -> 2 bytes
  g1[1] = (int)((td0 & 0xFFFFu) << 16);         // tensor_dim0[15:0]
  g1[2] = (int)((td0 >> 16) | (64u << 16));     // td0[31:16] | tensor_dim1=64
  g1[3] = (int)(32u << 16);                     // td1 hi=0 | tile_dim0=32
  g1[4] = 64;                                   // tile_dim1=64, tile_dim2=0
  g1[5] = K;                                    // tensor_dim0_stride[31:0]
  g1[6] = 0;                                    // stride0 hi | stride1 lo
  g1[7] = 0;                                    // stride1 hi
  v4i z4 = {0, 0, 0, 0};
  v8i z8 = {0, 0, 0, 0, 0, 0, 0, 0};
  __builtin_amdgcn_tensor_load_to_lds(g0, g1, z4, z4, z8, 0);
}

// ---------------------------------------------------------------------------
// f32 -> bf16 elementwise convert
__global__ void k_cvt(const float* __restrict__ src, bf16_t* __restrict__ dst,
                      int n) {
  int i = blockIdx.x * blockDim.x + threadIdx.x;
  if (i < n) dst[i] = (bf16_t)src[i];
}

// f32 [K,N] -> bf16 [N,K] transpose-convert (weights, B-operand layout)
__global__ void k_tcvt(const float* __restrict__ w, bf16_t* __restrict__ wt,
                       int K, int N) {
  int i = blockIdx.x * blockDim.x + threadIdx.x;
  if (i >= K * N) return;
  int k = i / N;
  int n = i - k * N;
  wt[(size_t)n * K + k] = (bf16_t)w[i];
}

// ---------------------------------------------------------------------------
// C[M,N] f32 = A[M,K] bf16 (row major) x BT[N,K] bf16 (stored transposed).
// Block = 8 waves = 128(M) x 64(N) tile; the shared 64x32 B tile is staged
// into LDS by the Tensor Data Mover (double buffered, s_wait_tensorcnt).
// Grid: (M/128, N/64); M,N multiples of 128/64 by construction.
__global__ void k_gemm(const bf16_t* __restrict__ A,
                       const bf16_t* __restrict__ BT,
                       float* __restrict__ C, int M, int N, int K) {
  __shared__ bf16_t Bs[2][64][32];  // 8 KB double-buffered B tile

  const int lane = threadIdx.x & 31;
  const int wave = threadIdx.x >> 5;
  const int ln   = lane & 15;
  const int hi   = lane >> 4;
  const int kbA  = hi * 16;   // A frag: K = kbA..+15
  const int kbB  = hi * 16;   // B frag: K = kbB..+15
  const int tm   = blockIdx.x;   // 128-row block
  const int tn   = blockIdx.y;   // 64-col block

  const bf16_t* ap = A + (size_t)(tm * 128 + wave * 16 + ln) * K;

  if (wave == 0)
    tdm_load_b(BT, tn * 64, 0, K, (unsigned)(size_t)&Bs[0][0][0]);

  v8f acc[4] = {};
  for (int k0 = 0; k0 < K; k0 += 32) {
    const int cur = (k0 >> 5) & 1;
    if (wave == 0) __builtin_amdgcn_s_wait_tensorcnt(0);
    __syncthreads();  // cur buffer ready; prev buffer fully consumed
    if (wave == 0 && k0 + 32 < K)
      tdm_load_b(BT, tn * 64, k0 + 32, K,
                 (unsigned)(size_t)&Bs[cur ^ 1][0][0]);

    __builtin_prefetch(ap + k0 + 128, 0, 1);  // global_prefetch_b8 (A stream)
    v16bf af = frag16(ap + k0 + kbA, ap + k0 + kbA + 8);
#pragma unroll
    for (int j = 0; j < 4; ++j) {
      const bf16_t* bp = &Bs[cur][j * 16 + ln][kbB];
      v16bf bfr = frag16(bp, bp + 8);   // ds_load_b128 x2
      acc[j] = wmma_bf16(af, bfr, acc[j]);
    }
  }

  const int crow = tm * 128 + wave * 16 + hi * 8;  // C: row = hi*8+e, col = ln
#pragma unroll
  for (int j = 0; j < 4; ++j)
#pragma unroll
    for (int e = 0; e < 8; ++e)
      C[(size_t)(crow + e) * N + tn * 64 + j * 16 + ln] = acc[j][e];
}

// ---------------------------------------------------------------------------
// Hierarchical RoPE: theta = base^(-(d&15)/16); d<32 local, d>=32 global.
// out[d<32]  = x[d]*cos - x[d+32]*sin
// out[d>=32] = x[d]*cos + x[d-32]*sin
__device__ __forceinline__ float hier_rope(const float* row, int t, int d) {
  const int fi = d & 15;
  const bool glob = d >= 32;
  const float logb = glob ? 13.1223633774f /*ln 5e5*/ : 9.2103403720f /*ln 1e4*/;
  const float theta = __expf(-(float)fi * 0.0625f * logb);
  const float ang = (float)t * theta;
  const float c = cosf(ang);
  const float s = sinf(ang);
  const float x0 = row[d];
  const float xp = glob ? row[d - 32] : row[d + 32];
  return glob ? fmaf(xp, s, x0 * c) : fmaf(-xp, s, x0 * c);
}

// qf [T, NH*64] f32 -> Qb bf16 [NH][T][64] with RoPE applied
__global__ void k_rope_q(const float* __restrict__ qf, bf16_t* __restrict__ Qb) {
  int i = blockIdx.x * blockDim.x + threadIdx.x;
  if (i >= T_SEQ * DMODEL) return;
  const int t = i >> 11;
  const int idx = i & (DMODEL - 1);
  const int d = idx & 63;
  const int h = idx >> 6;
  const float* row = qf + (size_t)t * DMODEL + h * 64;
  Qb[(size_t)h * T_SEQ * HDIM + (size_t)t * HDIM + d] =
      (bf16_t)hier_rope(row, t, d);
}

// kf/vf [T, NKV*64] f32 -> Kb bf16 [NKV][T][64] (RoPE), Vt bf16 [NKV][64][T]
__global__ void k_rope_kv(const float* __restrict__ kf,
                          const float* __restrict__ vf,
                          bf16_t* __restrict__ Kb, bf16_t* __restrict__ Vt) {
  int i = blockIdx.x * blockDim.x + threadIdx.x;
  if (i >= T_SEQ * NKV * HDIM) return;
  const int t = i >> 9;
  const int idx = i & 511;
  const int d = idx & 63;
  const int h = idx >> 6;
  const float* row = kf + (size_t)t * (NKV * HDIM) + h * 64;
  Kb[(size_t)h * T_SEQ * HDIM + (size_t)t * HDIM + d] =
      (bf16_t)hier_rope(row, t, d);
  Vt[(size_t)h * HDIM * T_SEQ + (size_t)d * T_SEQ + t] = (bf16_t)vf[i];
}

// ---------------------------------------------------------------------------
// Flash attention: 1 wave = (head h, 16-query tile). Online softmax with
// row stats held per-lane (C layout: 8 rows/lane, shfl_xor over 16 lanes).
// P re-shaped C-layout -> A-layout through LDS (same-wave DS is in-order,
// no block barrier => safe with divergent causal trip counts).
__global__ void k_attn(const bf16_t* __restrict__ Qb,
                       const bf16_t* __restrict__ Kb,
                       const bf16_t* __restrict__ Vt,
                       bf16_t* __restrict__ Ob) {
  __shared__ bf16_t Pl[8][16][32];  // 8 waves/block, 16x32 bf16 P tile each

  const int lane = threadIdx.x & 31;
  const int wave = threadIdx.x >> 5;
  const int gid  = blockIdx.x * (blockDim.x >> 5) + wave;
  const int h  = gid >> 7;     // 128 query tiles per head
  const int qt = gid & 127;
  const int q0 = qt * 16;
  const int ln  = lane & 15;
  const int hi  = lane >> 4;
  const int kbA = hi * 16;
  const int kbB = hi * 16;

  const bf16_t* Qh = Qb + (size_t)h * T_SEQ * HDIM;
  const bf16_t* Kh = Kb + (size_t)(h >> 2) * T_SEQ * HDIM;  // REP = 4
  const bf16_t* Vh = Vt + (size_t)(h >> 2) * HDIM * T_SEQ;

  // Q A-fragments for hd chunks [0,32) and [32,64)
  const bf16_t* qrow = Qh + (size_t)(q0 + ln) * HDIM;
  const v16bf qa0 = frag16(qrow + kbA,      qrow + kbA + 8);
  const v16bf qa1 = frag16(qrow + 32 + kbA, qrow + 32 + kbA + 8);

  v8f oacc[4] = {};
  float m_r[8], l_r[8];
#pragma unroll
  for (int e = 0; e < 8; ++e) { m_r[e] = -1e30f; l_r[e] = 0.f; }
  const float scale = 0.125f;  // 1/sqrt(64)

  for (int kb0 = 0; kb0 < q0 + 16; kb0 += 32) {
    float pv[2][8];
    float bmax[8];
#pragma unroll
    for (int e = 0; e < 8; ++e) bmax[e] = -3.0e38f;

#pragma unroll
    for (int s = 0; s < 2; ++s) {
      const int key = kb0 + s * 16 + ln;
      const int keyc = key < T_SEQ ? key : T_SEQ - 1;  // clamp frontier OOB
      const bf16_t* krow = Kh + (size_t)keyc * HDIM;   // B frag: contiguous hd
      v16bf kf0 = frag16(krow + kbB,      krow + kbB + 8);
      v16bf kf1 = frag16(krow + 32 + kbB, krow + 32 + kbB + 8);
      v8f sa = {};
      sa = wmma_bf16(qa0, kf0, sa);
      sa = wmma_bf16(qa1, kf1, sa);
#pragma unroll
      for (int e = 0; e < 8; ++e) {
        const int r = q0 + hi * 8 + e;
        float sv = sa[e] * scale;
        if (key > r) sv = -3.0e38f;  // causal mask
        pv[s][e] = sv;
        bmax[e] = fmaxf(bmax[e], sv);
      }
    }

    // row max across the 16 lanes of each half-wave
#pragma unroll
    for (int mk = 1; mk <= 8; mk <<= 1)
#pragma unroll
      for (int e = 0; e < 8; ++e)
        bmax[e] = fmaxf(bmax[e], __shfl_xor(bmax[e], mk, 32));

    float alpha[8], rsum[8];
#pragma unroll
    for (int e = 0; e < 8; ++e) {
      const float mn = fmaxf(m_r[e], bmax[e]);
      alpha[e] = __expf(m_r[e] - mn);
      m_r[e] = mn;
      rsum[e] = 0.f;
    }
#pragma unroll
    for (int s = 0; s < 2; ++s)
#pragma unroll
      for (int e = 0; e < 8; ++e) {
        const float p = __expf(pv[s][e] - m_r[e]);
        pv[s][e] = p;
        rsum[e] += p;
      }
#pragma unroll
    for (int mk = 1; mk <= 8; mk <<= 1)
#pragma unroll
      for (int e = 0; e < 8; ++e)
        rsum[e] += __shfl_xor(rsum[e], mk, 32);
#pragma unroll
    for (int e = 0; e < 8; ++e) l_r[e] = l_r[e] * alpha[e] + rsum[e];
#pragma unroll
    for (int j = 0; j < 4; ++j)
#pragma unroll
      for (int e = 0; e < 8; ++e) oacc[j][e] *= alpha[e];

    // P tile: C layout -> LDS -> A layout (same-wave DS ops are in-order)
#pragma unroll
    for (int s = 0; s < 2; ++s)
#pragma unroll
      for (int e = 0; e < 8; ++e)
        Pl[wave][hi * 8 + e][s * 16 + ln] = (bf16_t)pv[s][e];

    const v16bf pa = frag16(&Pl[wave][ln][kbA], &Pl[wave][ln][kbA + 8]);
#pragma unroll
    for (int j = 0; j < 4; ++j) {
      const bf16_t* vp = Vh + (size_t)(j * 16 + ln) * T_SEQ + kb0 + kbB;
      v16bf vb = frag16(vp, vp + 8);  // contiguous along keys (V transposed)
      oacc[j] = wmma_bf16(pa, vb, oacc[j]);
    }
  }

  // O / l, store bf16 into [T, NH*64] for the output projection
#pragma unroll
  for (int j = 0; j < 4; ++j)
#pragma unroll
    for (int e = 0; e < 8; ++e) {
      const float v = oacc[j][e] / l_r[e];
      const int r = q0 + hi * 8 + e;
      Ob[(size_t)r * DMODEL + h * 64 + j * 16 + ln] = (bf16_t)v;
    }
}

// ---------------------------------------------------------------------------
extern "C" void kernel_launch(void* const* d_in, const int* in_sizes, int n_in,
                              void* d_out, int out_size, void* d_ws,
                              size_t ws_size, hipStream_t stream) {
  const float* x  = (const float*)d_in[0];
  const float* wq = (const float*)d_in[1];
  const float* wk = (const float*)d_in[2];
  const float* wv = (const float*)d_in[3];
  const float* wo = (const float*)d_in[4];
  float* out = (float*)d_out;
  char* ws = (char*)d_ws;
  const size_t MB = 1024 * 1024;

  // Workspace map (72 MB total)
  bf16_t* xb  = (bf16_t*)(ws + 0);        //  8 MB  x  bf16 [T,D]
  bf16_t* wqT = (bf16_t*)(ws + 8  * MB);  //  8 MB  wq^T bf16 [2048,2048]
  bf16_t* wkT = (bf16_t*)(ws + 16 * MB);  //  2 MB  wk^T bf16 [512,2048]
  bf16_t* wvT = (bf16_t*)(ws + 18 * MB);  //  2 MB  wv^T bf16 [512,2048]
  bf16_t* woT = (bf16_t*)(ws + 20 * MB);  //  8 MB  wo^T bf16 [2048,2048]
  float*  qf  = (float*) (ws + 28 * MB);  // 16 MB  q f32 [T,2048]
  float*  kf  = (float*) (ws + 44 * MB);  //  4 MB  k f32 [T,512]
  float*  vf  = (float*) (ws + 48 * MB);  //  4 MB  v f32 [T,512]
  bf16_t* Qb  = (bf16_t*)(ws + 52 * MB);  //  8 MB  Q bf16 [NH][T][64]
  bf16_t* Kb  = (bf16_t*)(ws + 60 * MB);  //  2 MB  K bf16 [NKV][T][64]
  bf16_t* Vt  = (bf16_t*)(ws + 62 * MB);  //  2 MB  V bf16 [NKV][64][T]
  bf16_t* Ob  = (bf16_t*)(ws + 64 * MB);  //  8 MB  O bf16 [T,2048]

  // 1) precision conversion (A-side row-major, B-side transposed)
  k_cvt <<<(T_SEQ * DMODEL + 255) / 256, 256, 0, stream>>>(x, xb, T_SEQ * DMODEL);
  k_tcvt<<<(DMODEL * DMODEL + 255) / 256, 256, 0, stream>>>(wq, wqT, DMODEL, DMODEL);
  k_tcvt<<<(DMODEL * 512   + 255) / 256, 256, 0, stream>>>(wk, wkT, DMODEL, 512);
  k_tcvt<<<(DMODEL * 512   + 255) / 256, 256, 0, stream>>>(wv, wvT, DMODEL, 512);
  k_tcvt<<<(DMODEL * DMODEL + 255) / 256, 256, 0, stream>>>(wo, woT, DMODEL, DMODEL);

  // 2) QKV projections (WMMA, TDM-staged B): grid (M/128, N/64), 8 waves
  k_gemm<<<dim3(16, 32), 256, 0, stream>>>(xb, wqT, qf, 2048, 2048, 2048);
  k_gemm<<<dim3(16, 8),  256, 0, stream>>>(xb, wkT, kf, 2048, 512, 2048);
  k_gemm<<<dim3(16, 8),  256, 0, stream>>>(xb, wvT, vf, 2048, 512, 2048);

  // 3) hierarchical RoPE + per-head packing (V transposed for P@V B-frags)
  k_rope_q <<<(T_SEQ * DMODEL + 255) / 256, 256, 0, stream>>>(qf, Qb);
  k_rope_kv<<<(T_SEQ * NKV * HDIM + 255) / 256, 256, 0, stream>>>(kf, vf, Kb, Vt);

  // 4) causal flash attention: 32 heads * 128 query tiles, 8 waves/block
  k_attn<<<(NH * 128) / 8, 256, 0, stream>>>(Qb, Kb, Vt, Ob);

  // 5) output projection (WMMA) straight into d_out (f32)
  k_gemm<<<dim3(16, 32), 256, 0, stream>>>(Ob, woT, out, 2048, 2048, 2048);
}
